// TernaryLinear_74474732912767
// MI455X (gfx1250) — matrix operand order repaired and checked by reading, not compiled, not run on hardware
//
#include <hip/hip_runtime.h>
#include <hip/hip_bf16.h>

typedef __attribute__((ext_vector_type(16))) _Float16 v16h;
typedef __attribute__((ext_vector_type(8)))  _Float16 v8h;
typedef __attribute__((ext_vector_type(4)))  _Float16 v4h;
typedef __attribute__((ext_vector_type(8)))  float    v8f;
typedef __attribute__((ext_vector_type(4)))  float    v4f;

#define M_DIM 8192   // B*S = 4*2048
#define N_DIM 4096   // D_out
#define K_DIM 4096   // D_in

// ---------------------------------------------------------------------------
// Kernel 1: per-row ternary quantization of W -> f16 {-1, 0, +1}
// thresh = 0.7 * mean(|w_row|)
// ---------------------------------------------------------------------------
__global__ __launch_bounds__(256) void quantize_w_kernel(
    const float* __restrict__ w, _Float16* __restrict__ wq)
{
    __shared__ float red[256];
    const int row = blockIdx.x;
    const float* wr = w + (size_t)row * K_DIM;

    float s = 0.0f;
    #pragma unroll
    for (int i = threadIdx.x * 4; i < K_DIM; i += 256 * 4) {
        v4f v = *(const v4f*)(wr + i);
        s += fabsf(v.x) + fabsf(v.y) + fabsf(v.z) + fabsf(v.w);
    }
    red[threadIdx.x] = s;
    __syncthreads();
    #pragma unroll
    for (int off = 128; off > 0; off >>= 1) {
        if (threadIdx.x < off) red[threadIdx.x] += red[threadIdx.x + off];
        __syncthreads();
    }
    const float thresh = 0.7f * red[0] * (1.0f / (float)K_DIM);

    _Float16* qr = wq + (size_t)row * K_DIM;
    #pragma unroll
    for (int i = threadIdx.x * 4; i < K_DIM; i += 256 * 4) {
        v4f v = *(const v4f*)(wr + i);
        v4h q;
        q.x = (_Float16)((v.x > thresh) ? 1.0f : ((v.x < -thresh) ? -1.0f : 0.0f));
        q.y = (_Float16)((v.y > thresh) ? 1.0f : ((v.y < -thresh) ? -1.0f : 0.0f));
        q.z = (_Float16)((v.z > thresh) ? 1.0f : ((v.z < -thresh) ? -1.0f : 0.0f));
        q.w = (_Float16)((v.w > thresh) ? 1.0f : ((v.w < -thresh) ? -1.0f : 0.0f));
        *(v4h*)(qr + i) = q;
    }
}

// ---------------------------------------------------------------------------
// Kernel 2: x f32 -> f16 (vectorized); M_DIM*K_DIM divisible by 1024
// ---------------------------------------------------------------------------
__global__ __launch_bounds__(256) void cvt_x_kernel(
    const float* __restrict__ x, _Float16* __restrict__ xh)
{
    const size_t i = ((size_t)blockIdx.x * 256 + threadIdx.x) * 4;
    v4f v = *(const v4f*)(x + i);
    v4h h;
    h.x = (_Float16)v.x; h.y = (_Float16)v.y;
    h.z = (_Float16)v.z; h.w = (_Float16)v.w;
    *(v4h*)(xh + i) = h;
}

// ---------------------------------------------------------------------------
// Kernel 3: WMMA GEMM  out[M,N] = xh[M,K] * wq[N,K]^T + bias
// Block = 256 threads = 8 wave32 waves -> 128x256 output tile.
// Wave (wm in 0..1, wn in 0..3) -> 64x64 subtile = 4x4 WMMA 16x16 tiles.
// Per K-step (32): 16 global_load_b128 feed 16 v_wmma (1:1 ratio).
// ---------------------------------------------------------------------------
__global__ __launch_bounds__(256) void ternary_gemm_kernel(
    const _Float16* __restrict__ xh,   // [M, K]
    const _Float16* __restrict__ wq,   // [N, K]
    const float*    __restrict__ bias, // [N]
    float*          __restrict__ out)  // [M, N]
{
    const int lane = threadIdx.x & 31;
    const int wave = threadIdx.x >> 5;
    const int wm   = wave & 1;    // 0..1 -> M direction
    const int wn   = wave >> 1;   // 0..3 -> N direction
    const int lcol = lane & 15;   // row (A) / column (B,C,D) within 16
    const int lhi  = lane >> 4;   // half-wave select

    const int mb = blockIdx.y * 128 + wm * 64;
    const int nb = blockIdx.x * 256 + wn * 64;

    // A-fragment base pointers (contiguous k order, not the ISA table order):
    // lanes 0-15: K 0..15; lanes 16-31: K 16..31 (contiguous, matching B)
    const _Float16* aptr[4];
    aptr[0] = xh + (size_t)(mb + lcol) * K_DIM + (lhi ? 16 : 0);
    #pragma unroll
    for (int mi = 1; mi < 4; ++mi)
        aptr[mi] = aptr[mi - 1] + (size_t)16 * K_DIM;

    // B-fragment base pointers (contiguous k order, same as A):
    // lane holds column n = nb + ni*16 + lcol, 16 contiguous K values
    // (K 0..15 for lanes 0-15, K 16..31 for lanes 16-31)
    const _Float16* bptr[4];
    #pragma unroll
    for (int ni = 0; ni < 4; ++ni)
        bptr[ni] = wq + (size_t)(nb + ni * 16 + lcol) * K_DIM + (lhi ? 16 : 0);

    v8f acc[4][4] = {};

    for (int kb = 0; kb < K_DIM; kb += 32) {
        v16h a[4];
        #pragma unroll
        for (int mi = 0; mi < 4; ++mi) {
            v8h lo = *(const v8h*)(aptr[mi] + kb);
            v8h hi = *(const v8h*)(aptr[mi] + kb + 8);
            a[mi] = __builtin_shufflevector(lo, hi,
                      0,1,2,3,4,5,6,7,8,9,10,11,12,13,14,15);
        }
        v16h b[4];
        #pragma unroll
        for (int ni = 0; ni < 4; ++ni)
            b[ni] = *(const v16h*)(bptr[ni] + kb);

        #pragma unroll
        for (int mi = 0; mi < 4; ++mi)
            #pragma unroll
            for (int ni = 0; ni < 4; ++ni)
                acc[mi][ni] = __builtin_amdgcn_wmma_f32_16x16x32_f16(
                    /*neg_a=*/false, a[mi],
                    /*neg_b=*/false, b[ni],
                    /*c_mod=*/(short)0, acc[mi][ni],
                    /*reuse_a=*/false, /*reuse_b=*/false);
    }

    // Epilogue: add per-column bias, store f32.
    // C/D layout: VGPR r -> M = r + 8*lhi, lane%16 -> N column.
    float bv[4];
    #pragma unroll
    for (int ni = 0; ni < 4; ++ni)
        bv[ni] = bias[nb + ni * 16 + lcol];

    #pragma unroll
    for (int mi = 0; mi < 4; ++mi) {
        const int mrow = mb + mi * 16 + lhi * 8;
        #pragma unroll
        for (int ni = 0; ni < 4; ++ni) {
            const int ncol = nb + ni * 16 + lcol;
            #pragma unroll
            for (int r = 0; r < 8; ++r)
                out[(size_t)(mrow + r) * N_DIM + ncol] = acc[mi][ni][r] + bv[ni];
        }
    }
}

// ---------------------------------------------------------------------------
extern "C" void kernel_launch(void* const* d_in, const int* in_sizes, int n_in,
                              void* d_out, int out_size, void* d_ws, size_t ws_size,
                              hipStream_t stream)
{
    const float* x    = (const float*)d_in[0];  // [4, 2048, 4096]
    const float* w    = (const float*)d_in[1];  // [4096, 4096]
    const float* bias = (const float*)d_in[2];  // [4096]
    float* out        = (float*)d_out;          // [8192, 4096]

    // Workspace layout: wq (32 MiB f16) then xh (64 MiB f16)
    _Float16* wq = (_Float16*)d_ws;
    _Float16* xh = (_Float16*)((char*)d_ws + (size_t)N_DIM * K_DIM * sizeof(_Float16));

    quantize_w_kernel<<<N_DIM, 256, 0, stream>>>(w, wq);

    const size_t xelems = (size_t)M_DIM * K_DIM;          // 33,554,432
    cvt_x_kernel<<<(unsigned)(xelems / 1024), 256, 0, stream>>>(x, xh);

    dim3 grid(N_DIM / 256, M_DIM / 128);                  // (16, 64)
    ternary_gemm_kernel<<<grid, 256, 0, stream>>>(xh, wq, bias, out);
}
